// RekaTransformerBlock_16758962389125
// MI455X (gfx1250) — compile-verified, matrix operand order checked by reading, not run on hardware
//
#include <hip/hip_runtime.h>
#include <hip/hip_bf16.h>

// ---------------------------------------------------------------------------
// Reka transformer block for MI455X (gfx1250): wave32 + WMMA bf16 pipeline
// with async global->LDS staging (ASYNCcnt) and double-buffered GEMM tiles.
// ---------------------------------------------------------------------------

#define DEV static __device__ __forceinline__

typedef __attribute__((ext_vector_type(16))) __bf16 v16bf;
typedef __attribute__((ext_vector_type(8)))  __bf16 v8bf;
typedef __attribute__((ext_vector_type(8)))  float  v8f;
typedef __attribute__((ext_vector_type(4)))  int    v4i;

#define LDS_AS __attribute__((address_space(3)))
#define GLB_AS __attribute__((address_space(1)))

DEV v8f vzero8() {
    v8f z;
#pragma unroll
    for (int i = 0; i < 8; ++i) z[i] = 0.0f;
    return z;
}

// Async global -> LDS copy of one 16-byte chunk (tracked by ASYNCcnt).
DEV void async_b128(const __bf16* g, __bf16* l) {
#if __has_builtin(__builtin_amdgcn_global_load_async_to_lds_b128)
    __builtin_amdgcn_global_load_async_to_lds_b128(
        (GLB_AS v4i*)(uintptr_t)g,
        (LDS_AS v4i*)(uint32_t)(uintptr_t)l, 0, 0);
#else
    unsigned loff = (unsigned)(uintptr_t)l;
    asm volatile("global_load_async_to_lds_b128 %0, %1, off"
                 :: "v"(loff), "v"(g) : "memory");
#endif
}

DEV void wait_async0() {
#if __has_builtin(__builtin_amdgcn_s_wait_asynccnt)
    __builtin_amdgcn_s_wait_asynccnt(0);
#else
    asm volatile("s_wait_asynccnt 0" ::: "memory");
#endif
}

// Build a 16-element bf16 fragment from two 16-byte chunks (the CDNA5 A/B
// 16-bit layout: lane holds K = half*8..+7 and K = 16+half*8..+7).
DEV v16bf ld_frag(const __bf16* lo, const __bf16* hi) {
    v8bf a = *(const v8bf*)lo;
    v8bf b = *(const v8bf*)hi;
    return __builtin_shufflevector(a, b, 0, 1, 2, 3, 4, 5, 6, 7,
                                          8, 9, 10, 11, 12, 13, 14, 15);
}

DEV v8f wmma_bf16(v16bf a, v16bf b, v8f c) {
    return __builtin_amdgcn_wmma_f32_16x16x32_bf16(
        false, a, false, b, (short)0, c, false, false);
}

// ---------------------------------------------------------------------------
// Elementwise f32 -> bf16 convert (4 elems / thread)
// ---------------------------------------------------------------------------
__global__ __launch_bounds__(256)
void k_cvt_bf16(const float* __restrict__ in, __bf16* __restrict__ out) {
    size_t i = ((size_t)blockIdx.x * 256 + threadIdx.x) * 4;
    float4 v = *(const float4*)&in[i];
    union { __bf16 h[4]; uint2 u; } o;
    o.h[0] = (__bf16)v.x; o.h[1] = (__bf16)v.y;
    o.h[2] = (__bf16)v.z; o.h[3] = (__bf16)v.w;
    *(uint2*)&out[i] = o.u;
}

// ---------------------------------------------------------------------------
// Weight transpose + convert: W[K][N] f32 -> Wt[N][K] bf16 (32x32 LDS tiles)
// ---------------------------------------------------------------------------
__global__ __launch_bounds__(256)
void k_cvt_w_t(const float* __restrict__ W, __bf16* __restrict__ Wt,
               int K, int N) {
    __shared__ float t[32][33];
    const int k0 = blockIdx.y * 32, n0 = blockIdx.x * 32;
    const int tid = threadIdx.x;
    const int c = tid & 31, r0 = tid >> 5;
#pragma unroll
    for (int i = 0; i < 4; ++i) {
        int r = r0 + i * 8;
        t[r][c] = W[(size_t)(k0 + r) * N + n0 + c];
    }
    __syncthreads();
#pragma unroll
    for (int i = 0; i < 4; ++i) {
        int r = r0 + i * 8;
        Wt[(size_t)(n0 + r) * K + k0 + c] = (__bf16)t[c][r];
    }
}

// ---------------------------------------------------------------------------
// Tiled WMMA GEMM: C[M][N] f32 = A[M][K] bf16 * Bt[N][K] bf16 (+ residual)
// Block = 256 threads (8 waves), tile 128x128, K-step 64, double-buffered
// async global->LDS staging.
// ---------------------------------------------------------------------------
#define GKS 64
#define GKP 72  // padded LDS row stride (144 B, 16B-aligned)

__global__ __launch_bounds__(256)
void k_gemm_bf16(const __bf16* __restrict__ A, const __bf16* __restrict__ Bt,
                 const float* __restrict__ resid, float* __restrict__ C,
                 int M, int N, int K) {
    __shared__ __bf16 a_lds[2][128 * GKP];
    __shared__ __bf16 b_lds[2][128 * GKP];
    const int tid = threadIdx.x;
    const int lane = tid & 31, wave = tid >> 5;
    const int half = (lane >> 4) & 1, l16 = lane & 15;
    const int m_blk = blockIdx.y * 128, n_blk = blockIdx.x * 128;
    const int m0 = wave * 16;
    const int r_base = tid >> 3, c_st = (tid & 7) * 8;

    v8f acc[8];
#pragma unroll
    for (int i = 0; i < 8; ++i) acc[i] = vzero8();

    // Stage one 128x64 tile of A and Bt into LDS buffer `buf` (async).
    auto stage = [&](int k0, int buf) {
#pragma unroll
        for (int i = 0; i < 4; ++i) {
            int r = r_base + i * 32;
            async_b128(&A[(size_t)(m_blk + r) * K + k0 + c_st],
                       &a_lds[buf][r * GKP + c_st]);
            async_b128(&Bt[(size_t)(n_blk + r) * K + k0 + c_st],
                       &b_lds[buf][r * GKP + c_st]);
        }
    };

    const int T = K / GKS;
    stage(0, 0);
    for (int t = 0; t < T; ++t) {
        wait_async0();          // tile t landed (async loads retire in order)
        __syncthreads();        // visible to all waves; prior reads finished
        if (t + 1 < T) stage((t + 1) * GKS, (t + 1) & 1);  // overlap copy
        const __bf16* abase = &a_lds[t & 1][(m0 + l16) * GKP];
#pragma unroll
        for (int kk = 0; kk < 2; ++kk) {
            v16bf af = ld_frag(abase + kk * 32 + half * 8,
                               abase + kk * 32 + 16 + half * 8);
#pragma unroll
            for (int nt = 0; nt < 8; ++nt) {
                const __bf16* brow =
                    &b_lds[t & 1][(nt * 16 + l16) * GKP + kk * 32];
                acc[nt] = wmma_bf16(
                    af, ld_frag(brow + half * 8, brow + 16 + half * 8),
                    acc[nt]);
            }
        }
    }

#pragma unroll
    for (int nt = 0; nt < 8; ++nt) {
#pragma unroll
        for (int r = 0; r < 8; ++r) {
            int m = m_blk + m0 + half * 8 + r;
            int n = n_blk + nt * 16 + l16;
            float v = acc[nt][r];
            if (resid) v += resid[(size_t)m * N + n];
            C[(size_t)m * N + n] = v;
        }
    }
}

// ---------------------------------------------------------------------------
// Per-head RMSNorm over DH=128, f32 in -> bf16 out. One wave per row.
// ---------------------------------------------------------------------------
__global__ __launch_bounds__(256)
void k_rmsnorm(const float* __restrict__ in, const float* __restrict__ gamma,
               __bf16* __restrict__ out, int rows) {
    const int lane = threadIdx.x & 31, wave = threadIdx.x >> 5;
    const int row = blockIdx.x * 8 + wave;
    if (row >= rows) return;
    const float* p = in + (size_t)row * 128;
    float4 v = *(const float4*)&p[lane * 4];
    float ss = v.x * v.x + v.y * v.y + v.z * v.z + v.w * v.w;
#pragma unroll
    for (int m = 1; m < 32; m <<= 1) ss += __shfl_xor(ss, m, 32);
    float rs = rsqrtf(ss * (1.0f / 128.0f) + 1e-8f);
    float4 g = make_float4(1.f, 1.f, 1.f, 1.f);
    if (gamma) g = *(const float4*)&gamma[lane * 4];
    union { __bf16 h[4]; uint2 u; } o;
    o.h[0] = (__bf16)(v.x * rs * g.x);
    o.h[1] = (__bf16)(v.y * rs * g.y);
    o.h[2] = (__bf16)(v.z * rs * g.z);
    o.h[3] = (__bf16)(v.w * rs * g.w);
    *(uint2*)&out[(size_t)row * 128 + lane * 4] = o.u;
}

// ---------------------------------------------------------------------------
// Causal GQA flash attention. Grid: (q_tiles=16, H=16, B=2), block 256.
// Each of 8 waves owns 16 query rows. K staged async [key][dh]; V staged
// transposed [dh][key] through VGPRs (overlapping the async K copies).
// ---------------------------------------------------------------------------
#define AKP 136  // padded row stride (272 B, 16B-aligned)

__global__ __launch_bounds__(256)
void k_attn(const __bf16* __restrict__ qb, const __bf16* __restrict__ kb,
            const __bf16* __restrict__ vb, __bf16* __restrict__ ob) {
    __shared__ __bf16 k_lds[128 * AKP];
    __shared__ __bf16 v_lds[128 * AKP];
    __shared__ __bf16 p_lds[8][16][32];

    const int tid = threadIdx.x;
    const int lane = tid & 31, wave = tid >> 5;
    const int half = (lane >> 4) & 1, l16 = lane & 15;
    const int qt = blockIdx.x, h = blockIdx.y, b = blockIdx.z;
    const int kvh = h >> 2;            // GQA group of 4
    const int tok0 = b * 2048;
    const int q0 = qt * 128 + wave * 16;  // wave's first query row (in-batch)
    const float scale = 0.08838834764831845f;  // 128^-0.5

    // Q fragments: 16 rows x 128 dh, 4 K-chunks of 32.
    v16bf qf[4];
    {
        const int t = tok0 + q0 + l16;
        const __bf16* qrow = qb + ((size_t)t * 16 + h) * 128;
#pragma unroll
        for (int kk = 0; kk < 4; ++kk)
            qf[kk] = ld_frag(qrow + kk * 32 + half * 8,
                             qrow + kk * 32 + 16 + half * 8);
    }

    v8f o[8];
#pragma unroll
    for (int i = 0; i < 8; ++i) o[i] = vzero8();
    float mrow[8], lrow[8];  // replicated across the 16 lanes of each half
#pragma unroll
    for (int r = 0; r < 8; ++r) { mrow[r] = -3.0e38f; lrow[r] = 0.0f; }

    for (int kt = 0; kt <= qt; ++kt) {
        __syncthreads();  // previous tile fully consumed
        // --- async K tile: [key][dh], 128 keys x 16 chunks ---
#pragma unroll
        for (int i = 0; i < 8; ++i) {
            int idx = tid + i * 256;
            int key = idx >> 4, c = (idx & 15) * 8;
            int t = tok0 + kt * 128 + key;
            async_b128(&kb[((size_t)t * 4 + kvh) * 128 + c],
                       &k_lds[key * AKP + c]);
        }
        // --- V tile transposed through VGPRs (overlaps async K) ---
        uint4 uv[8];
#pragma unroll
        for (int i = 0; i < 8; ++i) {
            int idx = tid + i * 256;
            int key = idx >> 4, c = (idx & 15) * 8;
            int t = tok0 + kt * 128 + key;
            uv[i] = *(const uint4*)&vb[((size_t)t * 4 + kvh) * 128 + c];
        }
#pragma unroll
        for (int i = 0; i < 8; ++i) {
            int idx = tid + i * 256;
            int key = idx >> 4, c = (idx & 15) * 8;
            union { uint4 u; __bf16 hh[8]; } q;
            q.u = uv[i];
#pragma unroll
            for (int j = 0; j < 8; ++j)
                v_lds[(c + j) * AKP + key] = q.hh[j];
        }
        wait_async0();
        __syncthreads();

        // Skip fully-masked 32-key groups on the diagonal tile.
        const int gmax = (kt == qt) ? (wave * 16 + 15) / 32 : 3;
        for (int g = 0; g <= gmax; ++g) {
            // S = Q K^T for 32 keys (two 16-wide tiles), K-dim = dh = 128.
            v8f s0 = vzero8(), s1 = vzero8();
#pragma unroll
            for (int kk = 0; kk < 4; ++kk) {
                const __bf16* kr0 = &k_lds[(g * 32 + l16) * AKP + kk * 32];
                const __bf16* kr1 = &k_lds[(g * 32 + 16 + l16) * AKP + kk * 32];
                s0 = wmma_bf16(qf[kk],
                               ld_frag(kr0 + half * 8, kr0 + 16 + half * 8), s0);
                s1 = wmma_bf16(qf[kk],
                               ld_frag(kr1 + half * 8, kr1 + 16 + half * 8), s1);
            }
            // Scale + causal mask + per-row max over this 32-key group.
            const int qg = q0 + half * 8;          // + r
            const int kg0 = kt * 128 + g * 32 + l16;
            float p0[8], p1[8], tmax[8];
#pragma unroll
            for (int r = 0; r < 8; ++r) {
                float a = s0[r] * scale, c = s1[r] * scale;
                if (kg0 > qg + r) a = -3.0e38f;
                if (kg0 + 16 > qg + r) c = -3.0e38f;
                p0[r] = a; p1[r] = c;
                float t = fmaxf(a, c);
#pragma unroll
                for (int m = 1; m < 16; m <<= 1)
                    t = fmaxf(t, __shfl_xor(t, m, 32));
                tmax[r] = t;
            }
            // Online softmax update; stage P (bf16) re-laid-out for A-frag.
            float alpha[8];
#pragma unroll
            for (int r = 0; r < 8; ++r) {
                float mn = fmaxf(mrow[r], tmax[r]);
                bool live = mn > -1.0e37f;
                float al = live ? __expf(mrow[r] - mn) : 1.0f;
                float e0 = live ? __expf(p0[r] - mn) : 0.0f;
                float e1 = live ? __expf(p1[r] - mn) : 0.0f;
                float rsum = e0 + e1;
#pragma unroll
                for (int m = 1; m < 16; m <<= 1)
                    rsum += __shfl_xor(rsum, m, 32);
                lrow[r] = lrow[r] * al + rsum;
                mrow[r] = mn;
                alpha[r] = al;
                p_lds[wave][half * 8 + r][l16]      = (__bf16)e0;
                p_lds[wave][half * 8 + r][16 + l16] = (__bf16)e1;
            }
            // Rescale running output accumulators.
#pragma unroll
            for (int nt = 0; nt < 8; ++nt)
#pragma unroll
                for (int r = 0; r < 8; ++r) o[nt][r] *= alpha[r];
            // Same-wave LDS visibility for the staged P tile.
            asm volatile("s_wait_dscnt 0" ::: "memory");
            __builtin_amdgcn_wave_barrier();

            const __bf16* prow = &p_lds[wave][l16][0];
            v16bf pf = ld_frag(prow + half * 8, prow + 16 + half * 8);
#pragma unroll
            for (int nt = 0; nt < 8; ++nt) {
                const __bf16* vr = &v_lds[(nt * 16 + l16) * AKP + g * 32];
                o[nt] = wmma_bf16(pf, ld_frag(vr + half * 8, vr + 16 + half * 8),
                                  o[nt]);
            }
        }
    }

    // Finalize: divide by row sums, store bf16 [token][h*128 + dh].
#pragma unroll
    for (int r = 0; r < 8; ++r) lrow[r] = 1.0f / fmaxf(lrow[r], 1e-20f);
#pragma unroll
    for (int nt = 0; nt < 8; ++nt) {
#pragma unroll
        for (int r = 0; r < 8; ++r) {
            int t = tok0 + q0 + half * 8 + r;
            int dh = nt * 16 + l16;
            ob[((size_t)t * 16 + h) * 128 + dh] = (__bf16)(o[nt][r] * lrow[r]);
        }
    }
}

// ---------------------------------------------------------------------------
// Host-side orchestration
// ---------------------------------------------------------------------------
extern "C" void kernel_launch(void* const* d_in, const int* in_sizes, int n_in,
                              void* d_out, int out_size, void* d_ws, size_t ws_size,
                              hipStream_t stream) {
    (void)in_sizes; (void)n_in; (void)out_size; (void)ws_size;
    const float* x  = (const float*)d_in[0];
    const float* Wq = (const float*)d_in[1];
    const float* Wk = (const float*)d_in[2];
    const float* Wv = (const float*)d_in[3];
    const float* Wo = (const float*)d_in[4];
    const float* qg = (const float*)d_in[5];
    const float* kg = (const float*)d_in[6];
    float* out = (float*)d_out;

    const int M = 4096;        // tokens (2 batches x 2048)
    const int D = 2048;        // model dim
    const int KVD = 512;       // kv dim

    uint8_t* w = (uint8_t*)d_ws;
    __bf16* xb  = (__bf16*)(w);                // 16 MiB  x in bf16
    __bf16* wqt = (__bf16*)(w + 16777216);     //  8 MiB  Wq^T bf16
    __bf16* wkt = (__bf16*)(w + 25165824);     //  2 MiB  Wk^T bf16
    __bf16* wvt = (__bf16*)(w + 27262976);     //  2 MiB  Wv^T bf16
    __bf16* wot = (__bf16*)(w + 29360128);     //  8 MiB  Wo^T bf16
    float*  qf  = (float* )(w + 37748736);     // 32 MiB  Q f32 (later aliased)
    float*  kf  = (float* )(w + 71303168);     //  8 MiB  K f32
    float*  vf  = (float* )(w + 79691776);     //  8 MiB  V f32
    __bf16* qbf = (__bf16*)(w + 88080384);     // 16 MiB  Q bf16 (post-norm)
    __bf16* kbf = (__bf16*)(w + 104857600);    //  4 MiB  K bf16 (post-norm)
    __bf16* vbf = (__bf16*)(w + 109051904);    //  4 MiB  V bf16
    __bf16* aob = (__bf16*)(w + 37748736);     // alias over qf: attn out bf16
    // total workspace use: 113,246,208 bytes

    // 1) x -> bf16
    k_cvt_bf16<<<(M * D) / 1024, 256, 0, stream>>>(x, xb);
    // 2) weights -> transposed bf16
    k_cvt_w_t<<<dim3(D / 32, D / 32), 256, 0, stream>>>(Wq, wqt, D, D);
    k_cvt_w_t<<<dim3(KVD / 32, D / 32), 256, 0, stream>>>(Wk, wkt, D, KVD);
    k_cvt_w_t<<<dim3(KVD / 32, D / 32), 256, 0, stream>>>(Wv, wvt, D, KVD);
    k_cvt_w_t<<<dim3(D / 32, D / 32), 256, 0, stream>>>(Wo, wot, D, D);
    // 3) QKV projections (WMMA GEMMs)
    k_gemm_bf16<<<dim3(D / 128, M / 128), 256, 0, stream>>>(
        xb, wqt, nullptr, qf, M, D, D);
    k_gemm_bf16<<<dim3(KVD / 128, M / 128), 256, 0, stream>>>(
        xb, wkt, nullptr, kf, M, KVD, D);
    k_gemm_bf16<<<dim3(KVD / 128, M / 128), 256, 0, stream>>>(
        xb, wvt, nullptr, vf, M, KVD, D);
    // 4) QK RMSNorm (per head, DH=128) + V convert
    k_rmsnorm<<<(M * 16) / 8, 256, 0, stream>>>(qf, qg, qbf, M * 16);
    k_rmsnorm<<<(M * 4) / 8, 256, 0, stream>>>(kf, kg, kbf, M * 4);
    k_cvt_bf16<<<(M * KVD) / 1024, 256, 0, stream>>>(vf, vbf);
    // 5) causal GQA flash attention (WMMA)
    k_attn<<<dim3(16, 16, 2), 256, 0, stream>>>(qbf, kbf, vbf, aob);
    // 6) output projection + residual (WMMA GEMM)
    k_gemm_bf16<<<dim3(D / 128, M / 128), 256, 0, stream>>>(
        aob, wot, x, out, M, D, D);
}
